// MultiHeadAttention_37778532335657
// MI455X (gfx1250) — matrix operand order repaired and checked by reading, not compiled, not run on hardware
//
#include <hip/hip_runtime.h>
#include <hip/hip_bf16.h>

typedef __attribute__((ext_vector_type(16))) __bf16 v16bf;
typedef __attribute__((ext_vector_type(8)))  __bf16 v8bf;
typedef __attribute__((ext_vector_type(8)))  float  v8f;

#define B_SZ   4
#define S_LEN  2048
#define DM     1024
#define NH     16
#define DK     64
#define M_ROWS (B_SZ * S_LEN)   // 8192

#define WMMA_BF16(a, b, c) \
  __builtin_amdgcn_wmma_f32_16x16x32_bf16(false, (a), false, (b), (short)0, (c), false, false)

// ---- WMMA fragment loads from precomputed per-lane base pointers -----------
// A operand (16x32 MxK): base = &M[row0 + lane%16][ (lane/16)*16 ]; chunks at +k0 and +k0+8.
__device__ __forceinline__ v16bf fragA(const __bf16* __restrict__ pa, int k0) {
  v8bf lo = *(const v8bf*)(pa + k0);
  v8bf hi = *(const v8bf*)(pa + k0 + 8);
  v16bf r;
#pragma unroll
  for (int i = 0; i < 8; ++i) { r[i] = lo[i]; r[i + 8] = hi[i]; }
  return r;
}

// A operand from fp32 source (native casts -> v_cvt_pk_bf16_f32)
__device__ __forceinline__ v16bf fragA_f32(const float* __restrict__ pa, int k0) {
  float4 c0 = *(const float4*)(pa + k0);
  float4 c1 = *(const float4*)(pa + k0 + 4);
  float4 c2 = *(const float4*)(pa + k0 + 8);
  float4 c3 = *(const float4*)(pa + k0 + 12);
  v16bf r;
  r[0]  = (__bf16)c0.x; r[1]  = (__bf16)c0.y; r[2]  = (__bf16)c0.z; r[3]  = (__bf16)c0.w;
  r[4]  = (__bf16)c1.x; r[5]  = (__bf16)c1.y; r[6]  = (__bf16)c1.z; r[7]  = (__bf16)c1.w;
  r[8]  = (__bf16)c2.x; r[9]  = (__bf16)c2.y; r[10] = (__bf16)c2.z; r[11] = (__bf16)c2.w;
  r[12] = (__bf16)c3.x; r[13] = (__bf16)c3.y; r[14] = (__bf16)c3.z; r[15] = (__bf16)c3.w;
  return r;
}

// B operand (32x16 KxN): base = &Bt[col0 + lane%16][ (lane/16)*16 ]; one 32B chunk at +k0.
__device__ __forceinline__ v16bf fragB(const __bf16* __restrict__ pb, int k0) {
  return *(const v16bf*)(pb + k0);
}

// ---- 0) fp32 -> bf16 elementwise conversion (n multiple of 8) --------------
__global__ void cvt_kernel(const float* __restrict__ src, __bf16* __restrict__ dst, int n) {
  const int i = (blockIdx.x * blockDim.x + threadIdx.x) * 8;
  if (i >= n) return;
  float4 a = *(const float4*)(src + i);
  float4 b = *(const float4*)(src + i + 4);
  v8bf o;
  o[0] = (__bf16)a.x; o[1] = (__bf16)a.y; o[2] = (__bf16)a.z; o[3] = (__bf16)a.w;
  o[4] = (__bf16)b.x; o[5] = (__bf16)b.y; o[6] = (__bf16)b.z; o[7] = (__bf16)b.w;
  *(v8bf*)(dst + i) = o;
}

// ---- 1) Projections: Y = X @ W^T + b (all-bf16 operands) -------------------
// Wave tile 64x32 (MT=4, NT=2). Block 2x2 waves -> 128x64 tile.
// Store bf16 as [B,H,S,Dk] (Q,K) or transposed [B,H,Dk,S] (V).
template <bool TRANSPOSED_V>
__global__ __launch_bounds__(128, 1)
void proj_kernel(const __bf16* __restrict__ X, const __bf16* __restrict__ W,
                 const float* __restrict__ bias, __bf16* __restrict__ Y) {
  const int lane = threadIdx.x & 31;
  const int wave = threadIdx.x >> 5;
  const int m0 = blockIdx.x * 128 + (wave >> 1) * 64;
  const int n0 = blockIdx.y * 64 + (wave & 1) * 32;

  const __bf16* pa[4];
  const __bf16* pb[2];
#pragma unroll
  for (int i = 0; i < 4; ++i)
    pa[i] = X + (size_t)(m0 + i * 16 + (lane & 15)) * DM + ((lane >> 4) << 4);
#pragma unroll
  for (int j = 0; j < 2; ++j)
    pb[j] = W + (size_t)(n0 + j * 16 + (lane & 15)) * DM + ((lane >> 4) << 4);

  v8f acc[4][2] = {};
  for (int k0 = 0; k0 < DM; k0 += 32) {
    v16bf a[4], b[2];
#pragma unroll
    for (int i = 0; i < 4; ++i) a[i] = fragA(pa[i], k0);
#pragma unroll
    for (int j = 0; j < 2; ++j) b[j] = fragB(pb[j], k0);
#pragma unroll
    for (int i = 0; i < 4; ++i)
#pragma unroll
      for (int j = 0; j < 2; ++j)
        acc[i][j] = WMMA_BF16(a[i], b[j], acc[i][j]);
  }

  const int rbase = (lane >> 4) << 3;
#pragma unroll
  for (int ti = 0; ti < 4; ++ti)
#pragma unroll
    for (int tj = 0; tj < 2; ++tj) {
      const int col = n0 + tj * 16 + (lane & 15);
      const int h = col >> 6, d = col & 63;
      const float bv = bias[col];
#pragma unroll
      for (int i = 0; i < 8; ++i) {
        const int m = m0 + ti * 16 + rbase + i;
        const int b = m >> 11, s = m & 2047;
        const size_t idx = TRANSPOSED_V
            ? ((size_t)((b * NH + h) * DK + d)) * S_LEN + s
            : ((size_t)((b * NH + h) * S_LEN + s)) * DK + d;
        Y[idx] = (__bf16)(acc[ti][tj][i] + bv);
      }
    }
}

// ---- 2) Scores: P[bh,s,t] = (Qh[s,:].Kh[t,:]) / 8, fp32 into attn region ---
__global__ __launch_bounds__(128, 1)
void scores_kernel(const __bf16* __restrict__ Qh, const __bf16* __restrict__ Kh,
                   float* __restrict__ P) {
  const int lane = threadIdx.x & 31;
  const int wave = threadIdx.x >> 5;
  const int bh = blockIdx.z;
  const __bf16* Q = Qh + (size_t)bh * S_LEN * DK;
  const __bf16* K = Kh + (size_t)bh * S_LEN * DK;
  float* Pb = P + (size_t)bh * S_LEN * S_LEN;
  const int m0 = blockIdx.x * 128 + (wave >> 1) * 64;
  const int n0 = blockIdx.y * 64 + (wave & 1) * 32;

  const __bf16* pa[4];
  const __bf16* pb[2];
#pragma unroll
  for (int i = 0; i < 4; ++i)
    pa[i] = Q + (size_t)(m0 + i * 16 + (lane & 15)) * DK + ((lane >> 4) << 4);
#pragma unroll
  for (int j = 0; j < 2; ++j)
    pb[j] = K + (size_t)(n0 + j * 16 + (lane & 15)) * DK + ((lane >> 4) << 4);

  v8f acc[4][2] = {};
#pragma unroll
  for (int k0 = 0; k0 < DK; k0 += 32) {
    v16bf a[4], b[2];
#pragma unroll
    for (int i = 0; i < 4; ++i) a[i] = fragA(pa[i], k0);
#pragma unroll
    for (int j = 0; j < 2; ++j) b[j] = fragB(pb[j], k0);
#pragma unroll
    for (int i = 0; i < 4; ++i)
#pragma unroll
      for (int j = 0; j < 2; ++j)
        acc[i][j] = WMMA_BF16(a[i], b[j], acc[i][j]);
  }

  const int rbase = (lane >> 4) << 3;
#pragma unroll
  for (int ti = 0; ti < 4; ++ti)
#pragma unroll
    for (int tj = 0; tj < 2; ++tj) {
      const int col = n0 + tj * 16 + (lane & 15);
#pragma unroll
      for (int i = 0; i < 8; ++i) {
        const int row = m0 + ti * 16 + rbase + i;
        Pb[(size_t)row * S_LEN + col] = acc[ti][tj][i] * 0.125f;  // 1/sqrt(64)
      }
    }
}

// ---- 3) Row softmax in place (one 256-thread block per row of 2048) --------
__global__ void softmax_kernel(float* __restrict__ P) {
  __shared__ float red[256];
  float* row = P + (size_t)blockIdx.x * S_LEN;
  const int t = threadIdx.x;
  float m = -3.4e38f;
  for (int i = t; i < S_LEN; i += 256) m = fmaxf(m, row[i]);
  red[t] = m;
  __syncthreads();
  for (int s = 128; s > 0; s >>= 1) { if (t < s) red[t] = fmaxf(red[t], red[t + s]); __syncthreads(); }
  m = red[0];
  __syncthreads();
  float sum = 0.f;
  for (int i = t; i < S_LEN; i += 256) { float e = __expf(row[i] - m); row[i] = e; sum += e; }
  red[t] = sum;
  __syncthreads();
  for (int s = 128; s > 0; s >>= 1) { if (t < s) red[t] += red[t + s]; __syncthreads(); }
  const float inv = 1.0f / red[0];
  for (int i = t; i < S_LEN; i += 256) row[i] *= inv;
}

// ---- 4) PV: out_h[s,d] = sum_t P[s,t]*V[t,d]; V pre-transposed [Dk,S] ------
// Wave tile 32x64 (MT=2, NT=4; N total = Dk = 64). Store bf16 as [B,S,H*Dk].
__global__ __launch_bounds__(128, 1)
void pv_kernel(const float* __restrict__ P, const __bf16* __restrict__ Vt,
               __bf16* __restrict__ AO) {
  const int lane = threadIdx.x & 31;
  const int wave = threadIdx.x >> 5;
  const int bh = blockIdx.z;
  const float* Pb = P + (size_t)bh * S_LEN * S_LEN;
  const __bf16* Vb = Vt + (size_t)bh * DK * S_LEN;
  const int m0 = blockIdx.x * 128 + wave * 32;

  const float*  pa[2];
  const __bf16* pb[4];
#pragma unroll
  for (int i = 0; i < 2; ++i)
    pa[i] = Pb + (size_t)(m0 + i * 16 + (lane & 15)) * S_LEN + ((lane >> 4) << 4);
#pragma unroll
  for (int j = 0; j < 4; ++j)
    pb[j] = Vb + (size_t)(j * 16 + (lane & 15)) * S_LEN + ((lane >> 4) << 4);

  v8f acc[2][4] = {};
  for (int k0 = 0; k0 < S_LEN; k0 += 32) {
    v16bf a[2], b[4];
#pragma unroll
    for (int i = 0; i < 2; ++i) a[i] = fragA_f32(pa[i], k0);
#pragma unroll
    for (int j = 0; j < 4; ++j) b[j] = fragB(pb[j], k0);
#pragma unroll
    for (int i = 0; i < 2; ++i)
#pragma unroll
      for (int j = 0; j < 4; ++j)
        acc[i][j] = WMMA_BF16(a[i], b[j], acc[i][j]);
  }

  const int b = bh >> 4, h = bh & 15;
  const int rbase = (lane >> 4) << 3;
#pragma unroll
  for (int ti = 0; ti < 2; ++ti)
#pragma unroll
    for (int tj = 0; tj < 4; ++tj) {
      const int d = tj * 16 + (lane & 15);
#pragma unroll
      for (int i = 0; i < 8; ++i) {
        const int s = m0 + ti * 16 + rbase + i;
        AO[(((size_t)(b * S_LEN + s)) * NH + h) * DK + d] = (__bf16)acc[ti][tj][i];
      }
    }
}

// ---- 5) Output projection: Out = AO @ Wo^T + bo, fp32 output ---------------
__global__ __launch_bounds__(128, 1)
void oproj_kernel(const __bf16* __restrict__ AO, const __bf16* __restrict__ Wo,
                  const float* __restrict__ bo, float* __restrict__ Out) {
  const int lane = threadIdx.x & 31;
  const int wave = threadIdx.x >> 5;
  const int m0 = blockIdx.x * 128 + (wave >> 1) * 64;
  const int n0 = blockIdx.y * 64 + (wave & 1) * 32;

  const __bf16* pa[4];
  const __bf16* pb[2];
#pragma unroll
  for (int i = 0; i < 4; ++i)
    pa[i] = AO + (size_t)(m0 + i * 16 + (lane & 15)) * DM + ((lane >> 4) << 4);
#pragma unroll
  for (int j = 0; j < 2; ++j)
    pb[j] = Wo + (size_t)(n0 + j * 16 + (lane & 15)) * DM + ((lane >> 4) << 4);

  v8f acc[4][2] = {};
  for (int k0 = 0; k0 < DM; k0 += 32) {
    v16bf a[4], b[2];
#pragma unroll
    for (int i = 0; i < 4; ++i) a[i] = fragA(pa[i], k0);
#pragma unroll
    for (int j = 0; j < 2; ++j) b[j] = fragB(pb[j], k0);
#pragma unroll
    for (int i = 0; i < 4; ++i)
#pragma unroll
      for (int j = 0; j < 2; ++j)
        acc[i][j] = WMMA_BF16(a[i], b[j], acc[i][j]);
  }

  const int rbase = (lane >> 4) << 3;
#pragma unroll
  for (int ti = 0; ti < 4; ++ti)
#pragma unroll
    for (int tj = 0; tj < 2; ++tj) {
      const int col = n0 + tj * 16 + (lane & 15);
      const float bv = bo[col];
#pragma unroll
      for (int i = 0; i < 8; ++i) {
        const int m = m0 + ti * 16 + rbase + i;
        Out[(size_t)m * DM + col] = acc[ti][tj][i] + bv;
      }
    }
}

extern "C" void kernel_launch(void* const* d_in, const int* in_sizes, int n_in,
                              void* d_out, int out_size, void* d_ws, size_t ws_size,
                              hipStream_t stream) {
  (void)in_sizes; (void)n_in; (void)out_size; (void)ws_size;
  const float* q   = (const float*)d_in[0];
  const float* k   = (const float*)d_in[1];
  const float* v   = (const float*)d_in[2];
  const float* w_q = (const float*)d_in[3];
  const float* b_q = (const float*)d_in[4];
  const float* w_k = (const float*)d_in[5];
  const float* b_k = (const float*)d_in[6];
  const float* w_v = (const float*)d_in[7];
  const float* b_v = (const float*)d_in[8];
  const float* w_o = (const float*)d_in[9];
  const float* b_o = (const float*)d_in[10];

  float* out  = (float*)d_out;
  float* attn = out + (size_t)M_ROWS * DM;  // out first, then [B,H,S,S] attn

  const size_t MB = 1024 * 1024;
  char* ws = (char*)d_ws;
  __bf16* Qh  = (__bf16*)(ws);              // 16 MB  [B,H,S,Dk]
  __bf16* Kh  = (__bf16*)(ws + 16 * MB);    // 16 MB  [B,H,S,Dk]
  __bf16* Vt  = (__bf16*)(ws + 32 * MB);    // 16 MB  [B,H,Dk,S]
  __bf16* AO  = (__bf16*)(ws + 48 * MB);    // 16 MB  [B,S,H*Dk]
  __bf16* qb  = (__bf16*)(ws + 64 * MB);    // 16 MB  bf16 copies of inputs
  __bf16* kb  = (__bf16*)(ws + 80 * MB);
  __bf16* vb  = (__bf16*)(ws + 96 * MB);
  __bf16* wqb = (__bf16*)(ws + 112 * MB);   // 2 MB each: bf16 weights
  __bf16* wkb = (__bf16*)(ws + 114 * MB);
  __bf16* wvb = (__bf16*)(ws + 116 * MB);
  __bf16* wob = (__bf16*)(ws + 118 * MB);

  const int n_x = M_ROWS * DM;  // 8,388,608
  const int n_w = DM * DM;      // 1,048,576
  cvt_kernel<<<dim3(n_x / 8 / 256), dim3(256), 0, stream>>>(q,   qb,  n_x);
  cvt_kernel<<<dim3(n_x / 8 / 256), dim3(256), 0, stream>>>(k,   kb,  n_x);
  cvt_kernel<<<dim3(n_x / 8 / 256), dim3(256), 0, stream>>>(v,   vb,  n_x);
  cvt_kernel<<<dim3(n_w / 8 / 256), dim3(256), 0, stream>>>(w_q, wqb, n_w);
  cvt_kernel<<<dim3(n_w / 8 / 256), dim3(256), 0, stream>>>(w_k, wkb, n_w);
  cvt_kernel<<<dim3(n_w / 8 / 256), dim3(256), 0, stream>>>(w_v, wvb, n_w);
  cvt_kernel<<<dim3(n_w / 8 / 256), dim3(256), 0, stream>>>(w_o, wob, n_w);

  const dim3 blk(128);  // 4 waves
  proj_kernel<false><<<dim3(M_ROWS / 128, DM / 64), blk, 0, stream>>>(qb, wqb, b_q, Qh);
  proj_kernel<false><<<dim3(M_ROWS / 128, DM / 64), blk, 0, stream>>>(kb, wkb, b_k, Kh);
  proj_kernel<true ><<<dim3(M_ROWS / 128, DM / 64), blk, 0, stream>>>(vb, wvb, b_v, Vt);
  scores_kernel<<<dim3(S_LEN / 128, S_LEN / 64, B_SZ * NH), blk, 0, stream>>>(Qh, Kh, attn);
  softmax_kernel<<<dim3(B_SZ * NH * S_LEN), dim3(256), 0, stream>>>(attn);
  pv_kernel<<<dim3(S_LEN / 128, 1, B_SZ * NH), blk, 0, stream>>>(attn, Vt, AO);
  oproj_kernel<<<dim3(M_ROWS / 128, DM / 64), blk, 0, stream>>>(AO, wob, b_o, out);
}
